// GATBackbone_37838661878173
// MI455X (gfx1250) — matrix operand order repaired and checked by reading, not compiled, not run on hardware
//
#include <hip/hip_runtime.h>
#include <hip/hip_bf16.h>

// ---------------------------------------------------------------------------
// Types for CDNA5 WMMA (wave32, 16x16x32 bf16 -> f32)
// ---------------------------------------------------------------------------
typedef __attribute__((ext_vector_type(16))) __bf16       bf16x16;
typedef __attribute__((ext_vector_type(8)))  float        f32x8;
typedef __attribute__((ext_vector_type(8)))  unsigned int u32x8;

union FragCast { u32x8 u; bf16x16 b; };

__device__ __forceinline__ unsigned short f2bf(float f) {
  union { float f; unsigned int u; } v; v.f = f;
  unsigned int r = v.u + 0x7FFFu + ((v.u >> 16) & 1u);  // round-to-nearest-even
  return (unsigned short)(r >> 16);
}

#define LEAKY 0.2f
#define TS   132   // padded LDS stride (bf16 elems): 264B rows -> conflict-free
#define ROWS 64    // output rows per block

// ---------------------------------------------------------------------------
// GEMM: Hout[N,128] = X[N,128] @ W[128,128], bf16 WMMA, f32 accumulate.
// One block = 64 rows x 128 cols. 8 waves: wave w -> row-tile (w&3),
// column-half (w>>2) covering 4 col-tiles; 16 WMMAs/wave, A-frag reused 4x.
// ---------------------------------------------------------------------------
__global__ __launch_bounds__(256)
void gat_gemm_wmma(const float* __restrict__ X, const float* __restrict__ W,
                   float* __restrict__ Hout, int N) {
  __shared__ unsigned short sWt[128 * TS];   // transposed: [n][k]
  __shared__ unsigned short sX [ROWS * TS];  // [m][k]

  const int tid = threadIdx.x;

  // stage W (float4 reads, scatter-transpose to [n][k])
  for (int i = tid; i < (128 * 128) / 4; i += 256) {
    int idx = i << 2;
    int k = idx >> 7, n = idx & 127;
    float4 wv = ((const float4*)W)[i];
    sWt[(n + 0) * TS + k] = f2bf(wv.x);
    sWt[(n + 1) * TS + k] = f2bf(wv.y);
    sWt[(n + 2) * TS + k] = f2bf(wv.z);
    sWt[(n + 3) * TS + k] = f2bf(wv.w);
  }

  // stage X (float4 reads, packed 32-bit LDS writes)
  const int row0 = blockIdx.x * ROWS;
  for (int i = tid; i < (ROWS * 128) / 4; i += 256) {
    int idx = i << 2;
    int r = idx >> 7, c = idx & 127;
    int gr = row0 + r;
    float4 xv = (gr < N) ? ((const float4*)X)[((size_t)gr * 128 + c) >> 2]
                         : make_float4(0.f, 0.f, 0.f, 0.f);
    unsigned int p0 = (unsigned)f2bf(xv.x) | ((unsigned)f2bf(xv.y) << 16);
    unsigned int p1 = (unsigned)f2bf(xv.z) | ((unsigned)f2bf(xv.w) << 16);
    *(unsigned int*)&sX[r * TS + c]     = p0;
    *(unsigned int*)&sX[r * TS + c + 2] = p1;
  }
  __syncthreads();

  const int lane = tid & 31;
  const int wave = tid >> 5;
  const int m    = lane & 15;
  const int hi   = lane >> 4;
  const int rt   = wave & 3;   // row tile 0..3
  const int ch   = wave >> 1 & ~0; // (placeholder, fixed below)
  const int chalf = wave >> 2; // column half 0..1
  const int arow = rt * 16 + m;

  f32x8 acc[4] = {{}, {}, {}, {}};

  #pragma unroll
  for (int kb = 0; kb < 4; ++kb) {
    const int k0 = kb << 5;
    // A 16x32 bf16 fragment: lanes0-15 K={k0..k0+15}; lanes16-31 +16
    FragCast a;
    const unsigned short* ap = &sX[arow * TS + k0 + hi * 16];
    #pragma unroll
    for (int j = 0; j < 4; ++j) {
      a.u[j]     = *(const unsigned int*)(ap + 2 * j);
      a.u[j + 4] = *(const unsigned int*)(ap + 8 + 2 * j);
    }
    // 4 column tiles share this A fragment
    #pragma unroll
    for (int c = 0; c < 4; ++c) {
      const int nb = chalf * 64 + c * 16 + m;  // this lane's B column
      FragCast bfr;
      const unsigned short* bp = &sWt[nb * TS + k0 + hi * 16];
      #pragma unroll
      for (int j = 0; j < 8; ++j)
        bfr.u[j] = *(const unsigned int*)(bp + 2 * j);
      acc[c] = __builtin_amdgcn_wmma_f32_16x16x32_bf16(false, a.b, false, bfr.b,
                                                       (short)0, acc[c], false, false);
    }
  }

  // store: lane owns column (chalf*64 + c*16 + m); VGPR r is row rt*16 + hi*8 + r
  const bool full = (row0 + ROWS) <= N;
  const int  rbase = row0 + rt * 16 + hi * 8;
  (void)ch;
  #pragma unroll
  for (int c = 0; c < 4; ++c) {
    const int gcol = chalf * 64 + c * 16 + m;
    float* op = Hout + (size_t)rbase * 128 + gcol;
    if (full) {
      #pragma unroll
      for (int r = 0; r < 8; ++r) op[(size_t)r * 128] = acc[c][r];
    } else {
      #pragma unroll
      for (int r = 0; r < 8; ++r)
        if (rbase + r < N) op[(size_t)r * 128] = acc[c][r];
    }
  }
}

// ---------------------------------------------------------------------------
// Per-node attention logits: as[n] = h[n] . a_src ; ad[n] = h[n] . a_dst
// ---------------------------------------------------------------------------
__global__ __launch_bounds__(256)
void alphas_kernel(const float* __restrict__ H, const float* __restrict__ a_src,
                   const float* __restrict__ a_dst, float* __restrict__ AS,
                   float* __restrict__ AD, int N) {
  size_t gt  = (size_t)blockIdx.x * blockDim.x + threadIdx.x;
  int wid    = (int)(gt >> 5);
  int lane   = threadIdx.x & 31;
  if (wid >= N) return;
  float4 h = ((const float4*)(H + (size_t)wid * 128))[lane];
  float4 s = ((const float4*)a_src)[lane];
  float4 d = ((const float4*)a_dst)[lane];
  float ps = h.x * s.x + h.y * s.y + h.z * s.z + h.w * s.w;
  float pd = h.x * d.x + h.y * d.y + h.z * d.z + h.w * d.w;
  #pragma unroll
  for (int off = 16; off > 0; off >>= 1) {
    ps += __shfl_xor(ps, off, 32);
    pd += __shfl_xor(pd, off, 32);
  }
  if (lane == 0) { AS[wid] = ps; AD[wid] = pd; }
}

// ---------------------------------------------------------------------------
// Utility fill
// ---------------------------------------------------------------------------
__global__ __launch_bounds__(256)
void fill_u32(unsigned int* __restrict__ p, unsigned int v, size_t n) {
  size_t i = (size_t)blockIdx.x * blockDim.x + threadIdx.x;
  if (i < n) p[i] = v;
}

// ---------------------------------------------------------------------------
// Edge phase. Edges 0..E-1 from edge_index, E..E+N-1 are self loops.
// ---------------------------------------------------------------------------
__device__ __forceinline__ void atomicMaxF32(float* addr, float val) {
  if (val >= 0.0f) atomicMax((int*)addr, __float_as_int(val));
  else             atomicMin((unsigned int*)addr, __float_as_uint(val));
}

__device__ __forceinline__ float edge_logit(const float* AS, const float* AD,
                                            int s, int d) {
  float v = AS[s] + AD[d];
  return (v > 0.0f) ? v : LEAKY * v;
}

__global__ __launch_bounds__(256)
void edge_max_kernel(const int* __restrict__ src, const int* __restrict__ dst,
                     const float* __restrict__ AS, const float* __restrict__ AD,
                     float* __restrict__ M, int E, int N) {
  int e = blockIdx.x * blockDim.x + threadIdx.x;
  if (e >= E + N) return;
  int s = (e < E) ? src[e] : (e - E);
  int d = (e < E) ? dst[e] : (e - E);
  atomicMaxF32(&M[d], edge_logit(AS, AD, s, d));
}

__global__ __launch_bounds__(256)
void edge_expsum_kernel(const int* __restrict__ src, const int* __restrict__ dst,
                        const float* __restrict__ AS, const float* __restrict__ AD,
                        const float* __restrict__ M, float* __restrict__ Z,
                        float* __restrict__ EX, int E, int N) {
  int e = blockIdx.x * blockDim.x + threadIdx.x;
  if (e >= E + N) return;
  int s = (e < E) ? src[e] : (e - E);
  int d = (e < E) ? dst[e] : (e - E);
  float ex = __expf(edge_logit(AS, AD, s, d) - M[d]);
  EX[e] = ex;
  atomicAdd(&Z[d], ex);
}

// one wave per edge: agg[dst] += (ex/z[dst]) * h[src]
__global__ __launch_bounds__(256)
void edge_agg_kernel(const int* __restrict__ src, const int* __restrict__ dst,
                     const float* __restrict__ EX, const float* __restrict__ Z,
                     const float* __restrict__ H, float* __restrict__ AGG,
                     int E, int N) {
  size_t gt = (size_t)blockIdx.x * blockDim.x + threadIdx.x;
  int e     = (int)(gt >> 5);
  int lane  = threadIdx.x & 31;
  if (e >= E + N) return;
  int s = (e < E) ? src[e] : (e - E);
  int d = (e < E) ? dst[e] : (e - E);
  const float* hs = H + (size_t)s * 128;
  __builtin_prefetch(hs + lane * 4, 0, 1);      // global_prefetch of source row
  float w = EX[e] / Z[d];
  float4 h = ((const float4*)hs)[lane];
  float* o = AGG + (size_t)d * 128 + lane * 4;
  atomicAdd(o + 0, w * h.x);
  atomicAdd(o + 1, w * h.y);
  atomicAdd(o + 2, w * h.z);
  atomicAdd(o + 3, w * h.w);
}

// ---------------------------------------------------------------------------
// out[n,f] = agg[n,f] + bias[f]  (optional ReLU), in place
// ---------------------------------------------------------------------------
__global__ __launch_bounds__(256)
void bias_act_kernel(float* __restrict__ AGG, const float* __restrict__ bias,
                     int N, int do_relu) {
  size_t i = (size_t)blockIdx.x * blockDim.x + threadIdx.x;
  if (i >= (size_t)N * 128) return;
  float v = AGG[i] + bias[i & 127];
  if (do_relu) v = fmaxf(v, 0.0f);
  AGG[i] = v;
}

// ---------------------------------------------------------------------------
// Pooling
// ---------------------------------------------------------------------------
__global__ __launch_bounds__(256)
void count_kernel(const int* __restrict__ batch, float* __restrict__ CNT, int N) {
  int i = blockIdx.x * blockDim.x + threadIdx.x;
  if (i < N) atomicAdd(&CNT[batch[i]], 1.0f);
}

__global__ __launch_bounds__(256)
void pool_accum_kernel(const float* __restrict__ H2, const int* __restrict__ batch,
                       float* __restrict__ out, int N, int t, int T) {
  size_t gt = (size_t)blockIdx.x * blockDim.x + threadIdx.x;
  int n     = (int)(gt >> 5);
  int lane  = threadIdx.x & 31;
  if (n >= N) return;
  int b = batch[n];
  float4 h = ((const float4*)(H2 + (size_t)n * 128))[lane];
  float* o = out + ((size_t)b * T + t) * 128 + lane * 4;
  atomicAdd(o + 0, h.x);
  atomicAdd(o + 1, h.y);
  atomicAdd(o + 2, h.z);
  atomicAdd(o + 3, h.w);
}

__global__ __launch_bounds__(256)
void scale_out_kernel(float* __restrict__ out, const float* __restrict__ CNT,
                      int B, int T) {
  int i = blockIdx.x * blockDim.x + threadIdx.x;
  if (i >= B * T * 128) return;
  int b = i / (T * 128);
  out[i] /= CNT[b];
}

// ---------------------------------------------------------------------------
// Host launcher
// ---------------------------------------------------------------------------
extern "C" void kernel_launch(void* const* d_in, const int* in_sizes, int n_in,
                              void* d_out, int out_size, void* d_ws, size_t ws_size,
                              hipStream_t stream) {
  const float* x     = (const float*)d_in[0];
  const int*   ei    = (const int*)d_in[1];
  const int*   batch = (const int*)d_in[2];
  const float* W1    = (const float*)d_in[3];
  const float* asr1  = (const float*)d_in[4];
  const float* adt1  = (const float*)d_in[5];
  const float* b1    = (const float*)d_in[6];
  const float* W2    = (const float*)d_in[7];
  const float* asr2  = (const float*)d_in[8];
  const float* adt2  = (const float*)d_in[9];
  const float* b2    = (const float*)d_in[10];
  float* out = (float*)d_out;

  const int N = in_sizes[2];
  const int E = in_sizes[1] / 2;
  const int F = 128;
  const int T = in_sizes[0] / (N * F);
  const int B = 16;
  const int Etot = E + N;
  const int* src = ei;
  const int* dst = ei + E;

  // workspace carve-out (256B aligned)
  char* w = (char*)d_ws;
  auto alloc = [&](size_t bytes) -> char* {
    char* p = w;
    w += (bytes + 255) & ~(size_t)255;
    return p;
  };
  float* H   = (float*)alloc((size_t)N * F * 4);
  float* AGG = (float*)alloc((size_t)N * F * 4);
  float* AS  = (float*)alloc((size_t)N * 4);
  float* AD  = (float*)alloc((size_t)N * 4);
  float* M   = (float*)alloc((size_t)N * 4);
  float* Z   = (float*)alloc((size_t)N * 4);
  float* EX  = (float*)alloc((size_t)Etot * 4);
  float* CNT = (float*)alloc((size_t)B * 4);
  (void)ws_size; (void)n_in; (void)out_size;

  dim3 blk(256);
  auto g = [](size_t n) { return dim3((unsigned)((n + 255) / 256)); };
  const unsigned NEG_INF = 0xFF800000u;
  const int gemm_blocks = (N + ROWS - 1) / ROWS;

  fill_u32<<<g((size_t)B * T * F), blk, 0, stream>>>((unsigned*)out, 0u, (size_t)B * T * F);
  fill_u32<<<g(B), blk, 0, stream>>>((unsigned*)CNT, 0u, (size_t)B);
  count_kernel<<<g(N), blk, 0, stream>>>(batch, CNT, N);

  for (int t = 0; t < T; ++t) {
    const float* Xt = x + (size_t)t * N * F;

    // ---- layer 1 ----
    gat_gemm_wmma<<<gemm_blocks, blk, 0, stream>>>(Xt, W1, H, N);
    alphas_kernel<<<g((size_t)N * 32), blk, 0, stream>>>(H, asr1, adt1, AS, AD, N);
    fill_u32<<<g(N), blk, 0, stream>>>((unsigned*)M, NEG_INF, (size_t)N);
    fill_u32<<<g(N), blk, 0, stream>>>((unsigned*)Z, 0u, (size_t)N);
    fill_u32<<<g((size_t)N * F), blk, 0, stream>>>((unsigned*)AGG, 0u, (size_t)N * F);
    edge_max_kernel<<<g(Etot), blk, 0, stream>>>(src, dst, AS, AD, M, E, N);
    edge_expsum_kernel<<<g(Etot), blk, 0, stream>>>(src, dst, AS, AD, M, Z, EX, E, N);
    edge_agg_kernel<<<g((size_t)Etot * 32), blk, 0, stream>>>(src, dst, EX, Z, H, AGG, E, N);
    bias_act_kernel<<<g((size_t)N * F), blk, 0, stream>>>(AGG, b1, N, 1);

    // ---- layer 2 ----
    gat_gemm_wmma<<<gemm_blocks, blk, 0, stream>>>(AGG, W2, H, N);
    alphas_kernel<<<g((size_t)N * 32), blk, 0, stream>>>(H, asr2, adt2, AS, AD, N);
    fill_u32<<<g(N), blk, 0, stream>>>((unsigned*)M, NEG_INF, (size_t)N);
    fill_u32<<<g(N), blk, 0, stream>>>((unsigned*)Z, 0u, (size_t)N);
    fill_u32<<<g((size_t)N * F), blk, 0, stream>>>((unsigned*)AGG, 0u, (size_t)N * F);
    edge_max_kernel<<<g(Etot), blk, 0, stream>>>(src, dst, AS, AD, M, E, N);
    edge_expsum_kernel<<<g(Etot), blk, 0, stream>>>(src, dst, AS, AD, M, Z, EX, E, N);
    edge_agg_kernel<<<g((size_t)Etot * 32), blk, 0, stream>>>(src, dst, EX, Z, H, AGG, E, N);
    bias_act_kernel<<<g((size_t)N * F), blk, 0, stream>>>(AGG, b2, N, 0);

    // ---- mean pool into [B, T, O] ----
    pool_accum_kernel<<<g((size_t)N * 32), blk, 0, stream>>>(AGG, batch, out, N, t, T);
  }
  scale_out_kernel<<<g((size_t)B * T * F), blk, 0, stream>>>(out, CNT, B, T);
}
